// HybridContrastiveLoss_88038239633769
// MI455X (gfx1250) — matrix operand order repaired and checked by reading, not compiled, not run on hardware
//
#include <hip/hip_runtime.h>
#include <stdint.h>

typedef __attribute__((ext_vector_type(16))) _Float16 v16h;
typedef __attribute__((ext_vector_type(8)))  _Float16 v8h;
typedef __attribute__((ext_vector_type(8)))  float    v8f;

union H16 { v16h v; v8h h[2]; };

#define NROWS 32768
#define CDIM  256
#define MCOLS 24576
#define NTILES (MCOLS / 16)
#define TEMP_INV 10.0f
#define SPLIT 2048.0f
#define SPLIT_INV (1.0f/2048.0f)
#define LPAD 264   // halves per LDS column row: 256 + 8 pad (16B) -> conflict-free b128 reads

// ---------------------------------------------------------------------------
// Kernel 1: split memory f32 -> (hi, lo*2048) f16 planes in ws
// ---------------------------------------------------------------------------
__global__ __launch_bounds__(256) void convert_kernel(const float* __restrict__ mem,
                                                      _Float16* __restrict__ hi,
                                                      _Float16* __restrict__ lo)
{
    long i = ((long)blockIdx.x * 256 + threadIdx.x) * 8;
    const long total = (long)MCOLS * CDIM;
    if (i >= total) return;
    float4 x0 = *(const float4*)(mem + i);
    float4 x1 = *(const float4*)(mem + i + 4);
    float xs[8] = {x0.x, x0.y, x0.z, x0.w, x1.x, x1.y, x1.z, x1.w};
    v8h vh, vl;
#pragma unroll
    for (int j = 0; j < 8; ++j) {
        float x = xs[j];
        _Float16 h = (_Float16)x;
        vh[j] = h;
        vl[j] = (_Float16)((x - (float)h) * SPLIT);
    }
    *(v8h*)(hi + i) = vh;
    *(v8h*)(lo + i) = vl;
}

// ---------------------------------------------------------------------------
// Kernel 2: pos[i] = dot(feat1_i, feat2_i) / TEMP  (one wave per row)
// ---------------------------------------------------------------------------
__global__ __launch_bounds__(256) void pos_kernel(const float* __restrict__ f1,
                                                  const float* __restrict__ f2,
                                                  float* __restrict__ pos)
{
    int row  = (blockIdx.x * 256 + threadIdx.x) >> 5;
    int lane = threadIdx.x & 31;
    if (row >= NROWS) return;
    const float* a = f1 + (long)row * CDIM;
    const float* b = f2 + (long)row * CDIM;
    float acc = 0.f;
#pragma unroll
    for (int i = 0; i < 8; ++i) acc += a[lane + 32 * i] * b[lane + 32 * i];
#pragma unroll
    for (int off = 16; off; off >>= 1) acc += __shfl_xor(acc, off, 32);
    if (lane == 0) pos[row] = acc * TEMP_INV;
}

// ---------------------------------------------------------------------------
// Kernel 3: fused GEMM (split-f16 WMMA) + online unmasked-max / masked-exp-sum.
// Block = 256 thr (8 waves), each wave owns 16 rows; A frags resident in VGPRs.
// B tiles (16 cols x 256 K, hi+lo) double-buffered via async global->LDS:
// iteration ct prefetches tile ct+1, then waits ASYNCcnt<=4 so the (in-order)
// older 4 loads of tile ct are complete, and computes while the prefetch flies.
// ---------------------------------------------------------------------------
__global__ __launch_bounds__(256) void stats_kernel(const float* __restrict__ feat1,
                                                    const float* __restrict__ feat2,
                                                    const int* __restrict__ lab1,
                                                    const int* __restrict__ lab2,
                                                    const _Float16* __restrict__ mem_hi,
                                                    const _Float16* __restrict__ mem_lo,
                                                    const int* __restrict__ mlab,
                                                    float* __restrict__ Mstat,
                                                    float* __restrict__ Sstat)
{
    __shared__ _Float16 sh_hi[2][16 * LPAD];
    __shared__ _Float16 sh_lo[2][16 * LPAD];

    const int tid   = threadIdx.x;
    const int lane  = tid & 31;
    const int wave  = tid >> 5;
    const int grow0 = blockIdx.x * 128 + wave * 16;        // in [0, 2N)
    const int branch = (grow0 >= NROWS);
    const int row0   = branch ? (grow0 - NROWS) : grow0;
    const float* feat = branch ? feat2 : feat1;
    const int*   labs = branch ? lab2  : lab1;

    // ---- A fragments: 16 rows x 256 K, contiguous layout (lane=row, hi-lanes take +16)
    const int arow = row0 + (lane & 15);
    const int koff = (lane & 16);                          // 0 or 16
    v16h ahi[8], alo[8];
#pragma unroll
    for (int s = 0; s < 8; ++s) {
        const float* p1 = feat + (long)arow * CDIM + s * 32 + koff;
        const float* p2 = p1 + 8;
        float4 a0 = *(const float4*)p1; float4 a1 = *(const float4*)(p1 + 4);
        float4 b0 = *(const float4*)p2; float4 b1 = *(const float4*)(p2 + 4);
        float xs[16] = {a0.x,a0.y,a0.z,a0.w, a1.x,a1.y,a1.z,a1.w,
                        b0.x,b0.y,b0.z,b0.w, b1.x,b1.y,b1.z,b1.w};
#pragma unroll
        for (int i = 0; i < 16; ++i) {
            float x = xs[i];
            _Float16 h = (_Float16)x;
            ahi[s][i] = h;
            alo[s][i] = (_Float16)((x - (float)h) * SPLIT);
        }
    }

    // row labels for this lane's 8 C-layout rows
    int llab[8];
    const int rbase = row0 + ((lane >> 4) << 3);
#pragma unroll
    for (int r = 0; r < 8; ++r) llab[r] = labs[rbase + r];

    float rm[8], rs[8];
#pragma unroll
    for (int r = 0; r < 8; ++r) { rm[r] = -__builtin_inff(); rs[r] = 0.f; }

    // B staging: thread t loads col c = t>>4, halves [(t&15)*16, +16) (2x b128 per plane)
    const int scol = tid >> 4;
    const int skk  = (tid & 15) * 16;
    const unsigned dhi0 = (unsigned)(uintptr_t)&sh_hi[0][scol * LPAD + skk];
    const unsigned dhi1 = (unsigned)(uintptr_t)&sh_hi[1][scol * LPAD + skk];
    const unsigned dlo0 = (unsigned)(uintptr_t)&sh_lo[0][scol * LPAD + skk];
    const unsigned dlo1 = (unsigned)(uintptr_t)&sh_lo[1][scol * LPAD + skk];

    // B fragment base for this lane (col = lane&15; hi lanes take K+16)
    const _Float16* bbh0 = &sh_hi[0][(lane & 15) * LPAD + (lane & 16)];
    const _Float16* bbh1 = &sh_hi[1][(lane & 15) * LPAD + (lane & 16)];
    const _Float16* bbl0 = &sh_lo[0][(lane & 15) * LPAD + (lane & 16)];
    const _Float16* bbl1 = &sh_lo[1][(lane & 15) * LPAD + (lane & 16)];

    auto issue_tile = [&](unsigned dh, unsigned dl, unsigned gb) {
        asm volatile("global_load_async_to_lds_b128 %0, %1, %2"
                     :: "v"(dh), "v"(gb), "s"(mem_hi) : "memory");
        asm volatile("global_load_async_to_lds_b128 %0, %1, %2 offset:16"
                     :: "v"(dh), "v"(gb), "s"(mem_hi) : "memory");
        asm volatile("global_load_async_to_lds_b128 %0, %1, %2"
                     :: "v"(dl), "v"(gb), "s"(mem_lo) : "memory");
        asm volatile("global_load_async_to_lds_b128 %0, %1, %2 offset:16"
                     :: "v"(dl), "v"(gb), "s"(mem_lo) : "memory");
    };

    // prologue: tile 0 -> buffer 0
    issue_tile(dhi0, dlo0, (unsigned)((scol * CDIM + skk) * 2));

    for (int ct = 0; ct < NTILES; ++ct) {
        __syncthreads();   // all waves done reading the buffer we prefetch into
        if (ct + 1 < NTILES) {
            const unsigned gb = (unsigned)(((((ct + 1) * 16) + scol) * CDIM + skk) * 2);
            issue_tile((ct & 1) ? dhi0 : dhi1, (ct & 1) ? dlo0 : dlo1, gb);
            asm volatile("s_wait_asynccnt 4" ::: "memory");  // older 4 (tile ct) done
        } else {
            asm volatile("s_wait_asynccnt 0" ::: "memory");
        }
        __syncthreads();   // tile ct visible to all waves

        const _Float16* bb_hi = (ct & 1) ? bbh1 : bbh0;
        const _Float16* bb_lo = (ct & 1) ? bbl1 : bbl0;

        v8f acc_hi = {0.f,0.f,0.f,0.f,0.f,0.f,0.f,0.f};
        v8f acc_cr = {0.f,0.f,0.f,0.f,0.f,0.f,0.f,0.f};
#pragma unroll
        for (int s = 0; s < 8; ++s) {
            H16 bh, bl;
            bh.h[0] = *(const v8h*)(bb_hi + 32 * s);
            bh.h[1] = *(const v8h*)(bb_hi + 32 * s + 8);
            bl.h[0] = *(const v8h*)(bb_lo + 32 * s);
            bl.h[1] = *(const v8h*)(bb_lo + 32 * s + 8);
            acc_hi = __builtin_amdgcn_wmma_f32_16x16x32_f16(false, ahi[s], false, bh.v,
                                                            (short)0, acc_hi, false, false);
            acc_cr = __builtin_amdgcn_wmma_f32_16x16x32_f16(false, ahi[s], false, bl.v,
                                                            (short)0, acc_cr, false, false);
            acc_cr = __builtin_amdgcn_wmma_f32_16x16x32_f16(false, alo[s], false, bh.v,
                                                            (short)0, acc_cr, false, false);
        }

        const int clab = mlab[ct * 16 + (lane & 15)];
#pragma unroll
        for (int r = 0; r < 8; ++r) {
            float v  = (acc_hi[r] + acc_cr[r] * SPLIT_INV) * TEMP_INV;
            float nm = fmaxf(rm[r], v);                      // max over UNMASKED sim
            float e  = (llab[r] != clab) ? __expf(v - nm) : 0.f;
            rs[r] = rs[r] * __expf(rm[r] - nm) + e;          // masked exp-sum
            rm[r] = nm;
        }
    }

    // merge (max,sum) across the 16 column-lanes of each half-wave
#pragma unroll
    for (int off = 1; off < 16; off <<= 1) {
#pragma unroll
        for (int r = 0; r < 8; ++r) {
            float om = __shfl_xor(rm[r], off, 32);
            float os = __shfl_xor(rs[r], off, 32);
            float nm = fmaxf(rm[r], om);
            rs[r] = rs[r] * __expf(rm[r] - nm) + os * __expf(om - nm);
            rm[r] = nm;
        }
    }
    if ((lane & 15) == 0) {
        const int g = grow0 + ((lane >> 4) << 3);
#pragma unroll
        for (int r = 0; r < 8; ++r) { Mstat[g + r] = rm[r]; Sstat[g + r] = rs[r]; }
    }
}

// ---------------------------------------------------------------------------
// Kernel 4: combine pos with (M,S), apply masks, reduce to scalar loss
// ---------------------------------------------------------------------------
__global__ __launch_bounds__(256) void finalize_kernel(const float* __restrict__ pos,
                                                       const float* __restrict__ Mst,
                                                       const float* __restrict__ Sst,
                                                       const float* __restrict__ lg1,
                                                       const float* __restrict__ lg2,
                                                       float* __restrict__ out)
{
    const int tid = threadIdx.x, lane = tid & 31, wv = tid >> 5;
    float a1 = 0.f, c1 = 0.f, a2 = 0.f, c2 = 0.f;
    for (int i = tid; i < NROWS; i += 256) {
        float p  = pos[i];
        float l1 = lg1[i], l2 = lg2[i];
        if ((l2 > 0.7f) && (l1 < l2)) {
            float Mx = Mst[i], S = Sst[i];
            float mx = fmaxf(p, Mx);
            float sm = __expf(p - mx) + S * __expf(Mx - mx);
            a1 += -__logf(__expf(p - mx) / (sm + 1e-8f) + 1e-8f);
            c1 += 1.f;
        }
        if ((l1 > 0.7f) && (l2 < l1)) {
            float Mx = Mst[NROWS + i], S = Sst[NROWS + i];
            float mx = fmaxf(p, Mx);
            float sm = __expf(p - mx) + S * __expf(Mx - mx);
            a2 += -__logf(__expf(p - mx) / (sm + 1e-8f) + 1e-8f);
            c2 += 1.f;
        }
    }
#pragma unroll
    for (int off = 16; off; off >>= 1) {
        a1 += __shfl_xor(a1, off, 32); c1 += __shfl_xor(c1, off, 32);
        a2 += __shfl_xor(a2, off, 32); c2 += __shfl_xor(c2, off, 32);
    }
    __shared__ float red[8][4];
    if (lane == 0) { red[wv][0] = a1; red[wv][1] = c1; red[wv][2] = a2; red[wv][3] = c2; }
    __syncthreads();
    if (tid == 0) {
        float A1 = 0, C1 = 0, A2 = 0, C2 = 0;
        for (int w = 0; w < 8; ++w) {
            A1 += red[w][0]; C1 += red[w][1]; A2 += red[w][2]; C2 += red[w][3];
        }
        out[0] = A1 / (C1 + 1e-12f) + A2 / (C2 + 1e-12f);
    }
}

// ---------------------------------------------------------------------------
extern "C" void kernel_launch(void* const* d_in, const int* in_sizes, int n_in,
                              void* d_out, int out_size, void* d_ws, size_t ws_size,
                              hipStream_t stream)
{
    const float* f1   = (const float*)d_in[0];   // output_feat1 [N,C]
    const float* f2   = (const float*)d_in[1];   // output_feat2 [N,C]
    const int*   pl1  = (const int*)d_in[2];     // pseudo_label1 [N]
    const int*   pl2  = (const int*)d_in[3];     // pseudo_label2 [N]
    const float* lg1  = (const float*)d_in[4];   // pseudo_logits1 [N]
    const float* lg2  = (const float*)d_in[5];   // pseudo_logits2 [N]
    const float* mem  = (const float*)d_in[6];   // memory [M,C]
    const int*   mlab = (const int*)d_in[7];     // memory_labels [M]

    _Float16* mem_hi = (_Float16*)d_ws;
    _Float16* mem_lo = mem_hi + (size_t)MCOLS * CDIM;
    float*    pos    = (float*)(mem_lo + (size_t)MCOLS * CDIM);
    float*    Mst    = pos + NROWS;
    float*    Sst    = Mst + 2 * NROWS;

    convert_kernel<<<(MCOLS * CDIM / 8 + 255) / 256, 256, 0, stream>>>(mem, mem_hi, mem_lo);
    pos_kernel<<<NROWS / 8, 256, 0, stream>>>(f1, f2, pos);
    stats_kernel<<<(2 * NROWS) / 128, 256, 0, stream>>>(f1, f2, pl1, pl2,
                                                        mem_hi, mem_lo, mlab, Mst, Sst);
    finalize_kernel<<<1, 256, 0, stream>>>(pos, Mst, Sst, lg1, lg2, (float*)d_out);
}
